// ContextPythiaModel_54065048322255
// MI455X (gfx1250) — matrix operand order repaired and checked by reading, not compiled, not run on hardware
//
#include <hip/hip_runtime.h>
#include <hip/hip_bf16.h>

#define V_  50304
#define E_  512
#define C_  256
#define L_  6
#define H_  8
#define I_  1024
#define D_  32
#define ROT_ 8
#define B_  4
#define S_  1024
#define BS  (B_*S_)

typedef __attribute__((ext_vector_type(16))) __bf16 v16bf;
typedef __attribute__((ext_vector_type(8)))  __bf16 v8bf;
typedef __attribute__((ext_vector_type(8)))  float  v8f;
typedef int v4i32 __attribute__((vector_size(16)));

__device__ inline __bf16 f2bf(float f) {
  unsigned u = __builtin_bit_cast(unsigned, f);
  unsigned r = u + 0x7fffu + ((u >> 16) & 1u);
  unsigned short h = (unsigned short)(r >> 16);
  return __builtin_bit_cast(__bf16, h);
}

__device__ inline float gelu_f(float x) {
  float x3 = x * x * x;
  return 0.5f * x * (1.0f + tanhf(0.7978845608028654f * (x + 0.044715f * x3)));
}

// ---- gfx1250 async global->LDS copy (16B), with safe fallback ----
#if __has_builtin(__builtin_amdgcn_global_load_async_to_lds_b128)
#define HAVE_ASYNC_LDS 1
#else
#define HAVE_ASYNC_LDS 0
#endif

__device__ inline void cp16_async(const __bf16* __restrict__ g, __bf16* l) {
#if HAVE_ASYNC_LDS
  // Builtin takes (global v4i32*, lds v4i32*, imm offset, imm cpol).
  // LDS generic pointers: low 32 bits are the LDS offset (ISA aperture rule).
  __builtin_amdgcn_global_load_async_to_lds_b128(
      (__attribute__((address_space(1))) v4i32*)(unsigned long long)(void*)g,
      (__attribute__((address_space(3))) v4i32*)(unsigned)(unsigned long long)(void*)l,
      0, 0);
#else
  *(v8bf*)l = *(const v8bf*)g;
#endif
}

__device__ inline void wait_async0() {
#if HAVE_ASYNC_LDS
#if __has_builtin(__builtin_amdgcn_s_wait_asynccnt)
  __builtin_amdgcn_s_wait_asynccnt(0);
#else
  asm volatile("s_wait_asynccnt 0" ::: "memory");
#endif
#endif
}

__device__ inline float block_reduce_sum(float v, float* red) {
  int t = threadIdx.x;
  red[t] = v; __syncthreads();
  for (int s = 128; s > 0; s >>= 1) {
    if (t < s) red[t] += red[t + s];
    __syncthreads();
  }
  float r = red[0]; __syncthreads();
  return r;
}

// ---------------- fp32 -> bf16 bulk convert ----------------
__global__ void k_cvt_bf16(const float* __restrict__ in, __bf16* __restrict__ out, int n) {
  int i = blockIdx.x * 256 + threadIdx.x;
  if (i < n) out[i] = f2bf(in[i]);
}

// ---------------- embedding gather + LN over E=512 ----------------
__global__ void k_embed_ln(const int* __restrict__ ids, const float* __restrict__ embed_w,
                           const float* __restrict__ en_w, const float* __restrict__ en_b,
                           float* __restrict__ emb) {
  __shared__ float red[256];
  int row = blockIdx.x;
  int id = ids[row];
  const float* src = embed_w + (long long)id * E_;
  int t = threadIdx.x;
  float v0 = src[t], v1 = src[t + 256];
  float mean = block_reduce_sum(v0 + v1, red) * (1.0f / E_);
  float d0 = v0 - mean, d1 = v1 - mean;
  float var = block_reduce_sum(d0 * d0 + d1 * d1, red) * (1.0f / E_);
  float inv = rsqrtf(var + 1e-5f);
  float* dst = emb + (long long)row * E_;
  dst[t]       = d0 * inv * en_w[t]       + en_b[t];
  dst[t + 256] = d1 * inv * en_w[t + 256] + en_b[t + 256];
}

// ---------------- LN over C=256, bf16 output (feeds WMMA GEMMs) ----------------
__global__ void k_ln256_bf(const float* __restrict__ x, const float* __restrict__ w,
                           const float* __restrict__ b, __bf16* __restrict__ y) {
  __shared__ float red[256];
  int row = blockIdx.x, t = threadIdx.x;
  float v = x[(long long)row * C_ + t];
  float mean = block_reduce_sum(v, red) * (1.0f / C_);
  float d = v - mean;
  float var = block_reduce_sum(d * d, red) * (1.0f / C_);
  y[(long long)row * C_ + t] = f2bf(d * rsqrtf(var + 1e-5f) * w[t] + b[t]);
}

// ---------------- sequential context scan (one block per batch) ----------------
__global__ void k_scan(const float* __restrict__ prev_ctx, const float* __restrict__ emb,
                       const float* __restrict__ ctx_w, const float* __restrict__ ctx_b,
                       const float* __restrict__ cn_w, const float* __restrict__ cn_b,
                       float* __restrict__ x) {
  __shared__ float ctx[C_];
  __shared__ float red[256];
  int b = blockIdx.x, c = threadIdx.x;
  ctx[c] = prev_ctx[b * C_ + c];
  __syncthreads();
  const float* wrow = ctx_w + (long long)c * (C_ + E_);
  float wb = ctx_b[c], gw = cn_w[c], gb = cn_b[c];
  for (int t = 0; t < S_; ++t) {
    const float* e = emb + (long long)(b * S_ + t) * E_;
    float acc = wb;
    for (int j = 0; j < C_; ++j) acc = fmaf(wrow[j], ctx[j], acc);
    for (int j = 0; j < E_; ++j) acc = fmaf(wrow[C_ + j], e[j], acc);
    float g = gelu_f(acc);
    float mean = block_reduce_sum(g, red) * (1.0f / C_);
    float d = g - mean;
    float var = block_reduce_sum(d * d, red) * (1.0f / C_);
    float nc = d * rsqrtf(var + 1e-5f) * gw + gb;
    ctx[c] = nc;
    __syncthreads();
    x[(long long)(b * S_ + t) * C_ + c] = nc;
  }
}

// ---------------- WMMA bf16 GEMM: out[M,N] = A[M,K] @ W[N,K]^T (+bias,+res) ----------------
#define GB_M 128
#define GB_N 128
#define GB_K 32
#define LDT  40   // bf16 row stride in LDS: 80B, 16B-aligned, conflict-reducing

// fuse: bit0 = gelu, bit1 = bf16 output
__global__ __launch_bounds__(256)
void k_gemm_wmma(const __bf16* __restrict__ A, const __bf16* __restrict__ W,
                 const float* __restrict__ bias,
                 const float* __restrict__ resA, const float* __restrict__ resB,
                 void* __restrict__ outp, int M, int N, int K, int fuse) {
  __shared__ __bf16 lds_a[GB_M * LDT];
  __shared__ __bf16 lds_b[GB_N * LDT];
  int tid = threadIdx.x;
  int wave = tid >> 5, lane = tid & 31;
  int hi = lane >> 4, ln = lane & 15;
  int wm = wave >> 2, wn = wave & 3;      // 2 x 4 wave grid; wave tile 64x32
  int m0 = blockIdx.y * GB_M;
  int n0 = blockIdx.x * GB_N;

  v8f acc[4][2] = {};

  int lr = tid >> 1;                      // 0..127 (row in both tiles)
  int lc = (tid & 1) * 16;                // 0 or 16

  for (int k0 = 0; k0 < K; k0 += GB_K) {
    const __bf16* ap = A + (long long)(m0 + lr) * K + k0 + lc;
    const __bf16* bp = W + (long long)(n0 + lr) * K + k0 + lc;
    cp16_async(ap,     &lds_a[lr * LDT + lc]);
    cp16_async(ap + 8, &lds_a[lr * LDT + lc + 8]);
    cp16_async(bp,     &lds_b[lr * LDT + lc]);
    cp16_async(bp + 8, &lds_b[lr * LDT + lc + 8]);
    if (k0 + GB_K < K) __builtin_prefetch(bp + GB_K, 0, 3);
    wait_async0();
    __syncthreads();

    v16bf afr[4], bfr[2];
    #pragma unroll
    for (int mt = 0; mt < 4; ++mt) {
      int mrow = wm * 64 + mt * 16 + ln;
      #pragma unroll
      for (int r = 0; r < 8; ++r) {
        int k = hi * 16 + 2 * r;                               // A 16x32 bf16 layout
        afr[mt][2 * r]     = lds_a[mrow * LDT + k];
        afr[mt][2 * r + 1] = lds_a[mrow * LDT + k + 1];
      }
    }
    #pragma unroll
    for (int nt = 0; nt < 2; ++nt) {
      int nrow = wn * 32 + nt * 16 + ln;
      #pragma unroll
      for (int r = 0; r < 8; ++r) {
        int k = hi * 16 + 2 * r;                               // B 32x16 bf16 layout
        bfr[nt][2 * r]     = lds_b[nrow * LDT + k];
        bfr[nt][2 * r + 1] = lds_b[nrow * LDT + k + 1];
      }
    }
    #pragma unroll
    for (int mt = 0; mt < 4; ++mt)
      #pragma unroll
      for (int nt = 0; nt < 2; ++nt)
        acc[mt][nt] = __builtin_amdgcn_wmma_f32_16x16x32_bf16(
            false, afr[mt], false, bfr[nt], (short)0, acc[mt][nt], false, false);
    __syncthreads();
  }

  #pragma unroll
  for (int mt = 0; mt < 4; ++mt) {
    #pragma unroll
    for (int nt = 0; nt < 2; ++nt) {
      int col = n0 + wn * 32 + nt * 16 + ln;
      float bv = bias ? bias[col] : 0.0f;
      #pragma unroll
      for (int r = 0; r < 8; ++r) {
        int row = m0 + wm * 64 + mt * 16 + r + 8 * hi;         // C/D f32 layout
        long long idx = (long long)row * N + col;
        float v = acc[mt][nt][r] + bv;
        if (fuse & 1) v = gelu_f(v);
        if (resA) v += resA[idx];
        if (resB) v += resB[idx];
        if (fuse & 2) ((__bf16*)outp)[idx] = f2bf(v);
        else          ((float*) outp)[idx] = v;
      }
    }
  }
}

// ---------------- RoPE + QKV split to bf16 [b,h,s,d] ----------------
__global__ void k_rope_split(const float* __restrict__ qkv, __bf16* __restrict__ Qb,
                             __bf16* __restrict__ Kb, __bf16* __restrict__ Vb) {
  int row = blockIdx.x;
  int c = threadIdx.x;
  int s = row & (S_ - 1), b = row >> 10;
  int h = c >> 5, d = c & 31;
  const float* p = qkv + (long long)row * (3 * C_);
  float qv = p[c], kv = p[C_ + c], vv = p[2 * C_ + c];
  if (d < ROT_) {
    float inv = __powf(10000.0f, -(float)((d & 3) * 2) / (float)ROT_);
    float ang = (float)s * inv;
    float cs = __cosf(ang), sn = __sinf(ang);
    int dh = (d < 4) ? (d + 4) : (d - 4);
    float qh = p[c - d + dh];        float kh = p[C_ + c - d + dh];
    if (d < 4) { qh = -qh; kh = -kh; }
    qv = qv * cs + qh * sn;
    kv = kv * cs + kh * sn;
  }
  long long o = ((long long)(b * H_ + h) * S_ + s) * D_ + d;
  Qb[o] = f2bf(qv); Kb[o] = f2bf(kv); Vb[o] = f2bf(vv);
}

// ---------------- causal flash attention, WMMA, wave-local (16 q rows/wave) ----------------
__global__ __launch_bounds__(256)
void k_attn(const __bf16* __restrict__ Q, const __bf16* __restrict__ Kbuf,
            const __bf16* __restrict__ Vbuf, __bf16* __restrict__ O) {
  __shared__ __bf16 p_lds[8][16 * 32];
  int tid = threadIdx.x, wave = tid >> 5, lane = tid & 31;
  int hi = lane >> 4, ln = lane & 15;
  int bh = blockIdx.x >> 3;
  int qt = blockIdx.x & 7;
  int b = bh >> 3, h = bh & 7;
  int q0 = qt * 128 + wave * 16;
  const __bf16* qp = Q    + (long long)bh * S_ * D_;
  const __bf16* kp = Kbuf + (long long)bh * S_ * D_;
  const __bf16* vp = Vbuf + (long long)bh * S_ * D_;

  // Q fragment: k-contiguous per lane (k = 16*hi + s), same order as K -> two 16B loads
  v8bf q_lo = *(const v8bf*)&qp[(q0 + ln) * D_ + 16 * hi];
  v8bf q_hi = *(const v8bf*)&qp[(q0 + ln) * D_ + 16 * hi + 8];
  v16bf qf = __builtin_shufflevector(q_lo, q_hi, 0,1,2,3,4,5,6,7,8,9,10,11,12,13,14,15);

  v8f o0 = {}, o1 = {};
  float rm[8], rl[8];
  #pragma unroll
  for (int r = 0; r < 8; ++r) { rm[r] = -1e30f; rl[r] = 0.0f; }
  const float scale = 0.17677669529663687f;   // 1/sqrt(32)

  int qhi = q0 + 15;
  for (int kb = 0; kb <= qhi; kb += 32) {
    // K fragments: k-contiguous per lane (k = 16*hi + s), same order as Q -> one 32B load each
    v16bf kf0 = *(const v16bf*)&kp[(kb + ln) * D_ + hi * 16];
    v16bf kf1 = *(const v16bf*)&kp[(kb + 16 + ln) * D_ + hi * 16];
    v8f s0 = {}, s1 = {};
    s0 = __builtin_amdgcn_wmma_f32_16x16x32_bf16(false, qf, false, kf0, (short)0, s0, false, false);
    s1 = __builtin_amdgcn_wmma_f32_16x16x32_bf16(false, qf, false, kf1, (short)0, s1, false, false);

    #pragma unroll
    for (int r = 0; r < 8; ++r) {
      int row = q0 + r + 8 * hi;
      float v0s = s0[r] * scale; if (kb + ln > row)      v0s = -1e9f;
      float v1s = s1[r] * scale; if (kb + 16 + ln > row) v1s = -1e9f;
      float mx = fmaxf(v0s, v1s);
      for (int off = 1; off <= 8; off <<= 1) mx = fmaxf(mx, __shfl_xor(mx, off, 32));
      float mnew = fmaxf(rm[r], mx);
      float sc = __expf(rm[r] - mnew);
      float e0 = __expf(v0s - mnew), e1 = __expf(v1s - mnew);
      float sum = e0 + e1;
      for (int off = 1; off <= 8; off <<= 1) sum += __shfl_xor(sum, off, 32);
      rl[r] = rl[r] * sc + sum;
      rm[r] = mnew;
      o0[r] *= sc; o1[r] *= sc;
      p_lds[wave][(r + 8 * hi) * 32 + ln]      = f2bf(e0);
      p_lds[wave][(r + 8 * hi) * 32 + 16 + ln] = f2bf(e1);
    }
    asm volatile("s_wait_dscnt 0" ::: "memory");  // wave-local LDS RAW fence

    v8bf p_lo = *(const v8bf*)&p_lds[wave][ln * 32 + 16 * hi];
    v8bf p_hi = *(const v8bf*)&p_lds[wave][ln * 32 + 16 * hi + 8];
    v16bf pf = __builtin_shufflevector(p_lo, p_hi, 0,1,2,3,4,5,6,7,8,9,10,11,12,13,14,15);

    v16bf vf0, vf1;
    #pragma unroll
    for (int r = 0; r < 8; ++r) {
      int kr = kb + hi * 16 + 2 * r;
      vf0[2 * r]     = vp[(long long)kr * D_ + ln];
      vf0[2 * r + 1] = vp[(long long)(kr + 1) * D_ + ln];
      vf1[2 * r]     = vp[(long long)kr * D_ + 16 + ln];
      vf1[2 * r + 1] = vp[(long long)(kr + 1) * D_ + 16 + ln];
    }
    o0 = __builtin_amdgcn_wmma_f32_16x16x32_bf16(false, pf, false, vf0, (short)0, o0, false, false);
    o1 = __builtin_amdgcn_wmma_f32_16x16x32_bf16(false, pf, false, vf1, (short)0, o1, false, false);
  }

  #pragma unroll
  for (int r = 0; r < 8; ++r) {
    int row = q0 + r + 8 * hi;
    float inv = 1.0f / rl[r];
    long long base = (long long)(b * S_ + row) * C_ + h * D_;
    O[base + ln]      = f2bf(o0[r] * inv);
    O[base + 16 + ln] = f2bf(o1[r] * inv);
  }
}

extern "C" void kernel_launch(void* const* d_in, const int* in_sizes, int n_in,
                              void* d_out, int out_size, void* d_ws, size_t ws_size,
                              hipStream_t stream) {
  const int*   ids   = (const int*)  d_in[0];
  const float* prevc = (const float*)d_in[1];
  const float* embw  = (const float*)d_in[2];
  const float* enw   = (const float*)d_in[3];
  const float* enb   = (const float*)d_in[4];
  const float* ctxw  = (const float*)d_in[5];
  const float* ctxb  = (const float*)d_in[6];
  const float* cnw   = (const float*)d_in[7];
  const float* cnb   = (const float*)d_in[8];
  const float* ln1w  = (const float*)d_in[9];
  const float* ln1b  = (const float*)d_in[10];
  const float* ln2w  = (const float*)d_in[11];
  const float* ln2b  = (const float*)d_in[12];
  const float* qkvw  = (const float*)d_in[13];
  const float* qkvb  = (const float*)d_in[14];
  const float* aow   = (const float*)d_in[15];
  const float* aob   = (const float*)d_in[16];
  const float* fc1w  = (const float*)d_in[17];
  const float* fc1b  = (const float*)d_in[18];
  const float* fc2w  = (const float*)d_in[19];
  const float* fc2b  = (const float*)d_in[20];
  const float* flnw  = (const float*)d_in[21];
  const float* flnb  = (const float*)d_in[22];
  const float* outw  = (const float*)d_in[23];
  float* logits = (float*)d_out;

  char* ws = (char*)d_ws;
  size_t off = 0;
  auto alloc = [&](size_t bytes) -> void* {
    void* p = ws + off; off += (bytes + 255) & ~(size_t)255; return p;
  };
  float*  emb   = (float*) alloc((size_t)BS * E_ * 4);
  float*  x     = (float*) alloc((size_t)BS * C_ * 4);
  __bf16* tmp1  = (__bf16*)alloc((size_t)BS * C_ * 2);
  float*  qkv   = (float*) alloc((size_t)BS * 3 * C_ * 4);
  __bf16* Qb    = (__bf16*)alloc((size_t)BS * C_ * 2);
  __bf16* Kb    = (__bf16*)alloc((size_t)BS * C_ * 2);
  __bf16* Vb    = (__bf16*)alloc((size_t)BS * C_ * 2);
  __bf16* obuf  = (__bf16*)alloc((size_t)BS * C_ * 2);
  float*  attn  = (float*) alloc((size_t)BS * C_ * 4);
  __bf16* h1    = (__bf16*)alloc((size_t)BS * I_ * 2);
  __bf16* xf    = (__bf16*)alloc((size_t)BS * C_ * 2);
  __bf16* qkvwb = (__bf16*)alloc((size_t)L_ * 3 * C_ * C_ * 2);
  __bf16* aowb  = (__bf16*)alloc((size_t)L_ * C_ * C_ * 2);
  __bf16* fc1wb = (__bf16*)alloc((size_t)L_ * I_ * C_ * 2);
  __bf16* fc2wb = (__bf16*)alloc((size_t)L_ * C_ * I_ * 2);
  __bf16* outwb = (__bf16*)alloc((size_t)V_ * C_ * 2);
  (void)ws_size; (void)n_in; (void)in_sizes; (void)out_size;

  // one-time bf16 weight conversion (graph-replay deterministic)
  {
    int n;
    n = L_ * 3 * C_ * C_; k_cvt_bf16<<<(n + 255) / 256, 256, 0, stream>>>(qkvw, qkvwb, n);
    n = L_ * C_ * C_;     k_cvt_bf16<<<(n + 255) / 256, 256, 0, stream>>>(aow,  aowb,  n);
    n = L_ * I_ * C_;     k_cvt_bf16<<<(n + 255) / 256, 256, 0, stream>>>(fc1w, fc1wb, n);
    n = L_ * C_ * I_;     k_cvt_bf16<<<(n + 255) / 256, 256, 0, stream>>>(fc2w, fc2wb, n);
    n = V_ * C_;          k_cvt_bf16<<<(n + 255) / 256, 256, 0, stream>>>(outw, outwb, n);
  }

  k_embed_ln<<<BS, 256, 0, stream>>>(ids, embw, enw, enb, emb);
  k_scan<<<B_, 256, 0, stream>>>(prevc, emb, ctxw, ctxb, cnw, cnb, x);

  for (int l = 0; l < L_; ++l) {
    k_ln256_bf<<<BS, 256, 0, stream>>>(x, ln1w + l * C_, ln1b + l * C_, tmp1);
    dim3 g1(3 * C_ / GB_N, BS / GB_M);
    k_gemm_wmma<<<g1, 256, 0, stream>>>(tmp1, qkvwb + (long long)l * 3 * C_ * C_,
                                        qkvb + l * 3 * C_, nullptr, nullptr,
                                        qkv, BS, 3 * C_, C_, 0);
    k_rope_split<<<BS, 256, 0, stream>>>(qkv, Qb, Kb, Vb);
    k_attn<<<B_ * H_ * (S_ / 128), 256, 0, stream>>>(Qb, Kb, Vb, obuf);
    dim3 g2(C_ / GB_N, BS / GB_M);
    k_gemm_wmma<<<g2, 256, 0, stream>>>(obuf, aowb + (long long)l * C_ * C_,
                                        aob + l * C_, nullptr, nullptr,
                                        attn, BS, C_, C_, 0);
    k_ln256_bf<<<BS, 256, 0, stream>>>(x, ln2w + l * C_, ln2b + l * C_, tmp1);
    dim3 g3(I_ / GB_N, BS / GB_M);
    k_gemm_wmma<<<g3, 256, 0, stream>>>(tmp1, fc1wb + (long long)l * I_ * C_,
                                        fc1b + l * I_, nullptr, nullptr,
                                        h1, BS, I_, C_, 3 /*gelu|bf16*/);
    dim3 g4(C_ / GB_N, BS / GB_M);
    k_gemm_wmma<<<g4, 256, 0, stream>>>(h1, fc2wb + (long long)l * C_ * I_,
                                        fc2b + l * C_, x, attn,
                                        x, BS, C_, I_, 0);   // x = x + attn + mlp
  }

  k_ln256_bf<<<BS, 256, 0, stream>>>(x, flnw, flnb, xf);
  dim3 g5(V_ / GB_N, BS / GB_M);
  k_gemm_wmma<<<g5, 256, 0, stream>>>(xf, outwb, nullptr, nullptr, nullptr,
                                      logits, BS, V_, C_, 0);
}
